// Attention_46995532153449
// MI455X (gfx1250) — matrix operand order repaired and checked by reading, not compiled, not run on hardware
//
#include <hip/hip_runtime.h>

// ---------------------------------------------------------------------------
// MI455X (gfx1250) fused attention:
//   qkv GEMM (bf16 WMMA) -> head-softmax attention (flash-style over k,
//   V tiles staged via Tensor Data Mover DMA) -> projection GEMM + bias.
// Softmax is over the HEAD axis (faithful to reference), so no running
// renormalization over k is needed: per k-block, normalize across the 16
// heads via one LDS round-trip and accumulate P@V immediately.
// ---------------------------------------------------------------------------

typedef __bf16 bf16;
typedef __attribute__((ext_vector_type(16))) __bf16 v16bf;
typedef __attribute__((ext_vector_type(2)))  __bf16 bf16x2;
typedef __attribute__((ext_vector_type(8)))  float  v8f;
typedef __attribute__((ext_vector_type(4)))  unsigned int u32x4;
typedef __attribute__((ext_vector_type(8)))  int i32x8;
typedef __attribute__((ext_vector_type(4)))  int i32x4;

#define B_  2
#define T_  2048
#define D_  1024
#define H_  16
#define HD_ 64
#define QKVC (3 * D_)   // 3072, row stride of qkv workspace
#define MROWS (B_ * T_) // 4096
#define VROW 66         // LDS V-tile row stride (64 + 1 DWORD TDM pad)

// A-matrix (16x32 bf16) K index for VGPR-pair i (contiguous, matches B fills):
// lanes 0-15: V0..7 -> K=0..15 ; lanes 16-31: K=16..31
__device__ __forceinline__ int aK(int i, int lane) {
    return (lane & 16) + (i << 1);
}

__device__ __forceinline__ bf16x2 loadPair(const float* p) {
    float2 f = *(const float2*)p;
    bf16x2 r; r.x = (bf16)f.x; r.y = (bf16)f.y; return r;
}
__device__ __forceinline__ bf16x2 loadPair(const bf16* p) {
    return *(const bf16x2*)p;
}

// ---------------------------------------------------------------------------
// GEMM: C[M,N] = A[M,K] * Bw[K,N] (+ bias). A is f32 or bf16; C is bf16 or f32.
// Block: 256 threads = 8 waves; wave w owns rows [m0, m0+16), block owns 64 cols.
// B tile (32x64) staged to LDS as bf16, transposed [n][k], padded to stride 34.
// ---------------------------------------------------------------------------
template <typename TA, typename TC>
__global__ void __launch_bounds__(256)
gemm_kernel(const TA* __restrict__ A, const float* __restrict__ Bw,
            TC* __restrict__ C, const float* __restrict__ bias,
            int M, int N, int Kd)
{
    __shared__ bf16 ldsB[64][34];

    const int tid  = threadIdx.x;
    const int lane = tid & 31;
    const int w    = tid >> 5;
    const int m0   = blockIdx.y * 128 + w * 16;
    const int n0   = blockIdx.x * 64;
    const int m    = lane & 15;
    const int g    = lane >> 4;

    v8f acc[4] = {};

    for (int k0 = 0; k0 < Kd; k0 += 32) {
        // --- cooperative B tile stage: 256 thr x 8 floats = 32x64 ---
        {
            const int kk = tid >> 3;         // 0..31
            const int nn = (tid & 7) << 3;   // 0,8,..,56
            const float* src = Bw + (long)(k0 + kk) * N + n0 + nn;
            float4 f0 = *(const float4*)src;
            float4 f1 = *(const float4*)(src + 4);
            ldsB[nn + 0][kk] = (bf16)f0.x; ldsB[nn + 1][kk] = (bf16)f0.y;
            ldsB[nn + 2][kk] = (bf16)f0.z; ldsB[nn + 3][kk] = (bf16)f0.w;
            ldsB[nn + 4][kk] = (bf16)f1.x; ldsB[nn + 5][kk] = (bf16)f1.y;
            ldsB[nn + 6][kk] = (bf16)f1.z; ldsB[nn + 7][kk] = (bf16)f1.w;
        }
        __syncthreads();

        // --- A fragment (16x32), pairs contiguous in K ---
        v16bf a;
        const TA* rowA = A + (long)(m0 + m) * Kd + k0;
#pragma unroll
        for (int i = 0; i < 8; ++i) {
            bf16x2 p = loadPair(rowA + aK(i, lane));
            a[2 * i] = p.x; a[2 * i + 1] = p.y;
        }

        // --- 4 B fragments from LDS + WMMA ---
#pragma unroll
        for (int j = 0; j < 4; ++j) {
            v16bf bb;
            const int n = j * 16 + m;
#pragma unroll
            for (int i = 0; i < 8; ++i) {
                const int k = (g << 4) + (i << 1);
                bf16x2 p = *(const bf16x2*)&ldsB[n][k];
                bb[2 * i] = p.x; bb[2 * i + 1] = p.y;
            }
            acc[j] = __builtin_amdgcn_wmma_f32_16x16x32_bf16(
                false, a, false, bb, (short)0, acc[j], false, false);
        }
        __syncthreads();
    }

    // --- epilogue: C layout M = r + 8g, N = lane&15 ---
#pragma unroll
    for (int j = 0; j < 4; ++j) {
#pragma unroll
        for (int r = 0; r < 8; ++r) {
            const long row = m0 + r + (g << 3);
            const int  col = n0 + j * 16 + m;
            float v = acc[j][r];
            if (bias) v += bias[col];
            C[row * N + col] = (TC)v;
        }
    }
}

// ---------------------------------------------------------------------------
// TDM: DMA a 32-row x 64-col bf16 tile (row stride QKVC in memory) into LDS,
// with +1 DWORD padding per 128-byte row (LDS row stride = 66 bf16).
// Descriptor bitfields per CDNA5 ISA 8.3/8.4.
// ---------------------------------------------------------------------------
__device__ __forceinline__ void tdm_load_v_tile(const bf16* gsrc, unsigned lds_off)
{
    const unsigned long long ga = (unsigned long long)(const void*)gsrc;
    u32x4 g0;
    g0[0] = 0x1u;                                         // count=1, user mode
    g0[1] = lds_off;                                      // lds_addr (bytes)
    g0[2] = (unsigned)(ga & 0xFFFFFFFFu);                 // global_addr[31:0]
    g0[3] = (unsigned)((ga >> 32) & 0x1FFFFFFu)           // global_addr[56:32]
          | (2u << 30);                                   // type = 2 (image)
    i32x8 g1;
    g1[0] = (1 << 16)          // data_size = 1 -> 2 bytes
          | (1 << 20)          // pad_enable
          | (4 << 22);         // pad_interval code 4 -> every 32 DWORDs (128B)
                               // pad_amount code 0 -> 1 DWORD
    g1[1] = (64 << 16);        // tensor_dim0 = 64 (bits 79:48, low half)
    g1[2] = (MROWS << 16);     // tensor_dim1 = 4096 (bits 111:80, low half)
    g1[3] = (64 << 16);        // tile_dim0 = 64 (bits 127:112)
    g1[4] = 32;                // tile_dim1 = 32 (bits 143:128)
    g1[5] = QKVC;              // tensor_dim0_stride = 3072 (bits 207:160, low)
    g1[6] = 0;
    g1[7] = 0;
    i32x4 g2 = {0, 0, 0, 0};
    i32x4 g3 = {0, 0, 0, 0};
#if defined(__clang_major__) && __clang_major__ >= 23
    i32x8 g4 = {0, 0, 0, 0, 0, 0, 0, 0};
    __builtin_amdgcn_tensor_load_to_lds(g0, g1, g2, g3, g4, 0);
#else
    __builtin_amdgcn_tensor_load_to_lds(g0, g1, g2, g3, 0);
#endif
}

// ---------------------------------------------------------------------------
// Attention with softmax over heads. Block: 512 thr = 16 waves; wave h = head h.
// Per (b, 16-query tile): loop k in blocks of 32:
//   TDM issues V-tile DMA (overlaps with score math) ->
//   scores (2x 16x16 tiles via 2 WMMA each) -> LDS ->
//   cooperative softmax across the 16 heads (same (q,k) lives in the same
//   lane/slot for every head) -> probs as A-fragment ->
//   s_wait_tensorcnt -> 4 WMMA P@V from the LDS V tile.
// ---------------------------------------------------------------------------
__global__ void __launch_bounds__(512)
attn_kernel(const bf16* __restrict__ qkv, bf16* __restrict__ attn)
{
    __shared__ float sc[H_][16][33];          // [head][q][k-in-block], pad 33
    __shared__ bf16  vtile[H_][32 * VROW];    // per-head TDM-loaded V tile

    const int tid  = threadIdx.x;
    const int lane = tid & 31;
    const int h    = tid >> 5;            // wave index == head
    const int b    = blockIdx.y;
    const int q0   = blockIdx.x * 16;
    const int m    = lane & 15;
    const int g    = lane >> 4;
    const int hoff = h * HD_;

    const unsigned vtile_lds = (unsigned)(size_t)(&vtile[h][0]);

    // Resident Q fragments (16x64 split into two 16x32 A-fragments)
    v16bf qa0, qa1;
    {
        const bf16* rowQ = qkv + (long)(b * T_ + q0 + m) * QKVC + hoff;
#pragma unroll
        for (int i = 0; i < 8; ++i) {
            const int k = aK(i, lane);
            bf16x2 p0 = *(const bf16x2*)(rowQ + k);
            bf16x2 p1 = *(const bf16x2*)(rowQ + 32 + k);
            qa0[2 * i] = p0.x; qa0[2 * i + 1] = p0.y;
            qa1[2 * i] = p1.x; qa1[2 * i + 1] = p1.y;
        }
    }

    v8f acc[4] = {};   // attn_out accumulator, 16q x 64d

    for (int kb = 0; kb < T_; kb += 32) {
        // ---- kick off V-tile DMA for this k-block (per-wave, own head) ----
        tdm_load_v_tile(qkv + (long)(b * T_ + kb) * QKVC + 2 * D_ + hoff,
                        vtile_lds);

        // ---- prefetch next k-block's K rows while we compute ----
        if (kb + 32 < T_) {
            const bf16* nk =
                qkv + (long)(b * T_ + kb + 32 + m) * QKVC + D_ + hoff;
            __builtin_prefetch(nk, 0, 3);
        }

        // ---- scores: two 16x16 tiles over this 32-token k-block ----
#pragma unroll
        for (int kt = 0; kt < 2; ++kt) {
            const bf16* rowK =
                qkv + (long)(b * T_ + kb + kt * 16 + m) * QKVC + D_ + hoff;
            v16bf kf0, kf1;   // K^T B-fragments, d 0..31 and 32..63
#pragma unroll
            for (int i = 0; i < 8; ++i) {
                const int dk = (g << 4) + (i << 1);
                bf16x2 p0 = *(const bf16x2*)(rowK + dk);
                bf16x2 p1 = *(const bf16x2*)(rowK + 32 + dk);
                kf0[2 * i] = p0.x; kf0[2 * i + 1] = p0.y;
                kf1[2 * i] = p1.x; kf1[2 * i + 1] = p1.y;
            }
            v8f s = {};
            s = __builtin_amdgcn_wmma_f32_16x16x32_bf16(
                false, qa0, false, kf0, (short)0, s, false, false);
            s = __builtin_amdgcn_wmma_f32_16x16x32_bf16(
                false, qa1, false, kf1, (short)0, s, false, false);
#pragma unroll
            for (int r = 0; r < 8; ++r)
                sc[h][r + (g << 3)][kt * 16 + m] = s[r];
        }
        __syncthreads();

        // ---- softmax across heads: 512 threads <-> 16q x 32k entries ----
        {
            const int mm = tid >> 5;
            const int nn = tid & 31;
            float v[H_];
            float mx = -1e30f;
#pragma unroll
            for (int hh = 0; hh < H_; ++hh) {
                v[hh] = sc[hh][mm][nn] * 0.125f;   // SCALE = HD^-0.5
                mx = fmaxf(mx, v[hh]);
            }
            float den = 0.f;
#pragma unroll
            for (int hh = 0; hh < H_; ++hh) {
                v[hh] = __expf(v[hh] - mx);
                den += v[hh];
            }
            const float rden = 1.0f / den;
#pragma unroll
            for (int hh = 0; hh < H_; ++hh)
                sc[hh][mm][nn] = v[hh] * rden;     // store probs in place
        }
        __syncthreads();

        // ---- probs 16x32 as A-fragment (bf16) ----
        v16bf pf;
#pragma unroll
        for (int i = 0; i < 8; ++i) {
            const int k = aK(i, lane);
            pf[2 * i]     = (bf16)sc[h][m][k];
            pf[2 * i + 1] = (bf16)sc[h][m][k + 1];
        }

        // ---- wait for the V-tile DMA, then P @ V: 4 WMMAs ----
        __builtin_amdgcn_s_wait_tensorcnt(0);
#pragma unroll
        for (int j = 0; j < 4; ++j) {
            v16bf vb;
#pragma unroll
            for (int i = 0; i < 8; ++i) {
                const int kk = (g << 4) + (i << 1);
                const bf16* p = &vtile[h][kk * VROW + j * 16 + m];
                vb[2 * i]     = p[0];
                vb[2 * i + 1] = p[VROW];   // next token row (padded stride)
            }
            acc[j] = __builtin_amdgcn_wmma_f32_16x16x32_bf16(
                false, pf, false, vb, (short)0, acc[j], false, false);
        }
        __syncthreads();   // protect sc before next k-block overwrites it
    }

    // ---- epilogue: attn[b, t, h*64 + d] as bf16 ----
#pragma unroll
    for (int j = 0; j < 4; ++j) {
#pragma unroll
        for (int r = 0; r < 8; ++r) {
            const long row = (long)(b * T_ + q0 + r + (g << 3));
            attn[row * D_ + hoff + j * 16 + m] = (bf16)acc[j][r];
        }
    }
}

// ---------------------------------------------------------------------------
extern "C" void kernel_launch(void* const* d_in, const int* in_sizes, int n_in,
                              void* d_out, int out_size, void* d_ws, size_t ws_size,
                              hipStream_t stream)
{
    const float* x      = (const float*)d_in[0];
    const float* w_qkv  = (const float*)d_in[1];
    const float* w_proj = (const float*)d_in[2];
    const float* b_proj = (const float*)d_in[3];

    bf16* qkv  = (bf16*)d_ws;                       // 4096 x 3072 bf16 (24 MB)
    bf16* attn = qkv + (size_t)MROWS * QKVC;        // 4096 x 1024 bf16 (8 MB)

    // 1) qkv = x @ w_qkv          (f32 in -> bf16 ws)
    gemm_kernel<float, bf16>
        <<<dim3(QKVC / 64, MROWS / 128), 256, 0, stream>>>(
            x, w_qkv, qkv, nullptr, MROWS, QKVC, D_);

    // 2) head-softmax attention   (bf16 ws -> bf16 ws)
    attn_kernel<<<dim3(T_ / 16, B_), 512, 0, stream>>>(qkv, attn);

    // 3) out = attn @ w_proj + b  (bf16 ws -> f32 out)
    gemm_kernel<bf16, float>
        <<<dim3(D_ / 64, MROWS / 128), 256, 0, stream>>>(
            attn, w_proj, (float*)d_out, b_proj, MROWS, D_, D_);
}
